// Attention_82214263980554
// MI455X (gfx1250) — 16-bit and 8-bit matrix operand order repaired and checked by reading, compiled, not run on hardware
//
#include <hip/hip_runtime.h>
#include <stdint.h>

typedef __attribute__((ext_vector_type(16))) _Float16 v16h;
typedef __attribute__((ext_vector_type(8)))  float    v8f;
typedef __attribute__((ext_vector_type(8)))  int      v8i;

#define B_  2
#define T_  2048
#define C_  2048
#define H_  16
#define HD_ 128
#define M_  (B_*T_)

__global__ __launch_bounds__(256) void k_quant_rows(
    const float* __restrict__ in, int8_t* __restrict__ outq,
    float* __restrict__ scale, int cols)
{
  const int row = blockIdx.x;
  const float* r = in + (size_t)row * cols;
  float mx = 0.f;
  for (int c = threadIdx.x; c < cols; c += blockDim.x)
    mx = fmaxf(mx, fabsf(r[c]));
#pragma unroll
  for (int m = 16; m >= 1; m >>= 1)
    mx = fmaxf(mx, __shfl_xor(mx, m, 32));
  __shared__ float red[8];
  if ((threadIdx.x & 31) == 0) red[threadIdx.x >> 5] = mx;
  __syncthreads();
  mx = red[0];
#pragma unroll
  for (int i = 1; i < 8; ++i) mx = fmaxf(mx, red[i]);
  const float s = mx * (1.0f / 127.0f) + 1e-8f;
  if (threadIdx.x == 0) scale[row] = s;
  const float inv = 1.0f / s;
  int8_t* o = outq + (size_t)row * cols;
  for (int c = threadIdx.x; c < cols; c += blockDim.x) {
    float q = rintf(fminf(fmaxf(r[c] * inv, -127.f), 127.f));
    o[c] = (int8_t)q;
  }
}

__global__ __launch_bounds__(256) void k_gemm_q8(
    const int8_t* __restrict__ Aq, const float* __restrict__ sA,
    const int8_t* __restrict__ Bq, const float* __restrict__ sB,
    float* __restrict__ outp)
{
  constexpr int K = C_;
  constexpr int N = C_;
  __shared__ __align__(16) int8_t As[128 * 128];
  __shared__ __align__(16) int8_t Bs[32 * 128];
  const int tid = threadIdx.x;
  const int w   = tid >> 5;
  const int L   = tid & 31;
  const int h   = L >> 4;
  const int ln  = L & 15;
  const int m0  = blockIdx.x * 128;
  const int n0  = blockIdx.y * 32;

  v8i acc[2];
#pragma unroll
  for (int nt = 0; nt < 2; ++nt)
#pragma unroll
    for (int i = 0; i < 8; ++i) acc[nt][i] = 0;

  for (int kk = 0; kk < K; kk += 128) {
#pragma unroll
    for (int i = 0; i < 4; ++i) {
      int idx = tid + i * 256;
      int row = idx >> 3, seg = idx & 7;
      *(int4*)(As + row * 128 + seg * 16) =
          *(const int4*)(Aq + (size_t)(m0 + row) * K + kk + seg * 16);
    }
    {
      int row = tid >> 3, seg = tid & 7;
      *(int4*)(Bs + row * 128 + seg * 16) =
          *(const int4*)(Bq + (size_t)(n0 + row) * K + kk + seg * 16);
    }
    if (kk + 128 < K)
      __builtin_prefetch(Aq + (size_t)(m0 + (tid >> 1)) * K + kk + 128, 0, 3);
    __syncthreads();

#pragma unroll
    for (int kh2 = 0; kh2 < 2; ++kh2) {
      union { v8i v; int i[8]; } af;
      const int8_t* Ab = As + (16 * w + ln) * 128 + kh2 * 64;
#pragma unroll
      for (int j = 0; j < 8; ++j)
        af.i[j] = *(const int*)(Ab + ((j >> 1) * 16 + h * 8 + (j & 1) * 4));
#pragma unroll
      for (int nt = 0; nt < 2; ++nt) {
        union { v8i v; int i[8]; } bf;
        const int8_t* Bb = Bs + (nt * 16 + ln) * 128 + kh2 * 64;
#pragma unroll
        for (int j = 0; j < 8; ++j)
          bf.i[j] = *(const int*)(Bb + ((j >> 1) * 16 + h * 8 + (j & 1) * 4));
        acc[nt] = __builtin_amdgcn_wmma_i32_16x16x64_iu8(
            true, af.v, true, bf.v, acc[nt], false, false);
      }
    }
    __syncthreads();
  }

#pragma unroll
  for (int nt = 0; nt < 2; ++nt) {
    const float sb = sB[n0 + nt * 16 + ln];
#pragma unroll
    for (int r = 0; r < 8; ++r) {
      const int m = m0 + 16 * w + r + 8 * h;
      outp[(size_t)m * N + n0 + nt * 16 + ln] = (float)acc[nt][r] * sA[m] * sb;
    }
  }
}

__global__ __launch_bounds__(256) void k_norm_rope_cvt(
    const float* __restrict__ qlin, const float* __restrict__ klin,
    const float* __restrict__ vlin,
    const float* __restrict__ qg, const float* __restrict__ kg,
    const float* __restrict__ cosT, const float* __restrict__ sinT,
    _Float16* __restrict__ qo, _Float16* __restrict__ ko,
    _Float16* __restrict__ vo)
{
  const int tok = blockIdx.x;
  const int z = blockIdx.y;
  const int b = tok / T_;
  const int t = tok % T_;
  if (z == 2) {
    const float* row = vlin + (size_t)tok * C_;
    for (int c = threadIdx.x; c < C_; c += blockDim.x) {
      int hh = c >> 7, d = c & 127;
      vo[(((size_t)b * H_ + hh) * T_ + t) * HD_ + d] = (_Float16)row[c];
    }
    return;
  }
  const float* row = (z == 0) ? (qlin + (size_t)tok * C_)
                              : (klin + (size_t)tok * C_);
  const float* g = (z == 0) ? qg : kg;
  _Float16* out = (z == 0) ? qo : ko;

  float ss = 0.f;
  for (int c = threadIdx.x; c < C_; c += blockDim.x) {
    float v = row[c];
    ss += v * v;
  }
#pragma unroll
  for (int m = 16; m >= 1; m >>= 1) ss += __shfl_xor(ss, m, 32);
  __shared__ float red[8];
  if ((threadIdx.x & 31) == 0) red[threadIdx.x >> 5] = ss;
  __syncthreads();
  ss = 0.f;
#pragma unroll
  for (int i = 0; i < 8; ++i) ss += red[i];
  const float rs = rsqrtf(ss * (1.0f / C_) + 1e-6f);

  for (int p = threadIdx.x; p < (C_ / 2); p += blockDim.x) {
    int hh = p >> 6, i = p & 63;
    int c1 = hh * HD_ + 2 * i;
    float x1 = row[c1] * rs * g[c1];
    float x2 = row[c1 + 1] * rs * g[c1 + 1];
    float co = cosT[(size_t)t * (HD_ / 2) + i];
    float si = sinT[(size_t)t * (HD_ / 2) + i];
    _Float16* dst = out + (((size_t)b * H_ + hh) * T_ + t) * HD_;
    dst[2 * i]     = (_Float16)(x1 * co - x2 * si);
    dst[2 * i + 1] = (_Float16)(x1 * si + x2 * co);
  }
}

__global__ __launch_bounds__(128) void k_attn(
    const _Float16* __restrict__ qh, const _Float16* __restrict__ kh,
    const _Float16* __restrict__ vh, const int* __restrict__ maskp,
    float* __restrict__ outo)
{
  __shared__ __align__(16) _Float16 ksh[32 * 128];
  __shared__ __align__(16) _Float16 vsh[128 * 32];
  __shared__ __align__(16) _Float16 psh[4][16 * 32];
  const int tid = threadIdx.x;
  const int w   = tid >> 5;
  const int L   = tid & 31;
  const int h   = L >> 4;
  const int ln  = L & 15;
  const int hh  = blockIdx.y;
  const int b   = blockIdx.z;
  const int q0  = (blockIdx.x * 4 + w) * 16;
  const int nvalid = maskp[b];
  const size_t headbase = ((size_t)b * H_ + hh) * (size_t)T_ * HD_;

  union FragH { v16h v; unsigned u[8]; };

  FragH qf[4];
  {
    const _Float16* qrow = qh + headbase + (size_t)(q0 + ln) * HD_;
#pragma unroll
    for (int cq = 0; cq < 4; ++cq)
#pragma unroll
      for (int j = 0; j < 8; ++j) {
        int baseK = cq * 32 + (j >> 2) * 16 + h * 8 + (j & 3) * 2;
        qf[cq].u[j] = *(const unsigned*)(qrow + baseK);
      }
  }

  v8f oAcc[8];
#pragma unroll
  for (int c = 0; c < 8; ++c)
#pragma unroll
    for (int r = 0; r < 8; ++r) oAcc[c][r] = 0.f;
  float mrow[8], lrow[8];
#pragma unroll
  for (int r = 0; r < 8; ++r) { mrow[r] = -1e30f; lrow[r] = 0.f; }

  const float scale = 0.088388347648318447f;

  for (int kt = 0; kt < T_ / 32; ++kt) {
    const _Float16* kg = kh + headbase + (size_t)kt * 32 * HD_;
    const _Float16* vg = vh + headbase + (size_t)kt * 32 * HD_;
#pragma unroll
    for (int i = 0; i < 4; ++i)
      ((uint4*)ksh)[tid + i * 128] = ((const uint4*)kg)[tid + i * 128];
#pragma unroll
    for (int i = 0; i < 32; ++i) {
      int idx = tid + i * 128;
      int kk2 = idx >> 7, d = idx & 127;
      vsh[d * 32 + kk2] = vg[idx];
    }
    __syncthreads();

    v8f sc[2];
#pragma unroll
    for (int st = 0; st < 2; ++st) {
      v8f s;
#pragma unroll
      for (int r = 0; r < 8; ++r) s[r] = 0.f;
      const _Float16* kr = ksh + (st * 16 + ln) * 128;
#pragma unroll
      for (int cq = 0; cq < 4; ++cq) {
        FragH bf;
#pragma unroll
        for (int j = 0; j < 8; ++j) {
          int ks = cq * 32 + (j >> 2) * 16 + h * 8 + (j & 3) * 2;
          bf.u[j] = *(const unsigned*)(kr + ks);
        }
        s = __builtin_amdgcn_wmma_f32_16x16x32_f16(false, qf[cq].v, false,
                                                   bf.v, (short)0, s, false,
                                                   false);
      }
      const int kgidx = kt * 32 + st * 16 + ln;
      const bool valid = kgidx < nvalid;
#pragma unroll
      for (int r = 0; r < 8; ++r) s[r] = valid ? s[r] * scale : -1e9f;
      sc[st] = s;
    }

    _Float16* pw = &psh[w][0];
#pragma unroll
    for (int r = 0; r < 8; ++r) {
      float x = fmaxf(sc[0][r], sc[1][r]);
#pragma unroll
      for (int m = 1; m < 16; m <<= 1) x = fmaxf(x, __shfl_xor(x, m, 16));
      float mnew  = fmaxf(mrow[r], x);
      float alpha = __expf(mrow[r] - mnew);
      float p0 = __expf(sc[0][r] - mnew);
      float p1 = __expf(sc[1][r] - mnew);
      float rsum = p0 + p1;
#pragma unroll
      for (int m = 1; m < 16; m <<= 1) rsum += __shfl_xor(rsum, m, 16);
      lrow[r] = lrow[r] * alpha + rsum;
      mrow[r] = mnew;
#pragma unroll
      for (int c = 0; c < 8; ++c) oAcc[c][r] *= alpha;
      int prow = r + 8 * h;
      pw[prow * 32 + ln]      = (_Float16)p0;
      pw[prow * 32 + 16 + ln] = (_Float16)p1;
    }
    asm volatile("s_wait_dscnt 0" ::: "memory");

    FragH af;
#pragma unroll
    for (int j = 0; j < 8; ++j) {
      int baseK = (j >> 2) * 16 + h * 8 + (j & 3) * 2;
      af.u[j] = *(const unsigned*)(pw + ln * 32 + baseK);
    }
#pragma unroll
    for (int c = 0; c < 8; ++c) {
      FragH bf;
      const _Float16* vr = vsh + (c * 16 + ln) * 32;
#pragma unroll
      for (int j = 0; j < 8; ++j) {
        int ks = (j >> 2) * 16 + h * 8 + (j & 3) * 2;
        bf.u[j] = *(const unsigned*)(vr + ks);
      }
      oAcc[c] = __builtin_amdgcn_wmma_f32_16x16x32_f16(false, af.v, false,
                                                       bf.v, (short)0,
                                                       oAcc[c], false, false);
    }
    __syncthreads();
  }

#pragma unroll
  for (int r = 0; r < 8; ++r) {
    const int qrow = q0 + r + 8 * h;
    const float invl = 1.0f / lrow[r];
    float* orow = outo + ((size_t)b * T_ + qrow) * C_ + hh * HD_;
#pragma unroll
    for (int c = 0; c < 8; ++c) orow[c * 16 + ln] = oAcc[c][r] * invl;
  }
}

extern "C" void kernel_launch(void* const* d_in, const int* in_sizes, int n_in,
                              void* d_out, int out_size, void* d_ws,
                              size_t ws_size, hipStream_t stream)
{
  const float* x     = (const float*)d_in[0];
  const int*   maskp = (const int*)d_in[1];
  const float* wq    = (const float*)d_in[2];
  const float* wk    = (const float*)d_in[3];
  const float* wv    = (const float*)d_in[4];
  const float* wo    = (const float*)d_in[5];
  const float* qg    = (const float*)d_in[6];
  const float* kg    = (const float*)d_in[7];
  const float* cosT  = (const float*)d_in[8];
  const float* sinT  = (const float*)d_in[9];

  char* ws = (char*)d_ws;
  constexpr size_t XQ_OFF   = 0x0000000;
  constexpr size_t SX_OFF   = 0x0800000;
  constexpr size_t WQ8_OFF  = 0x0810000;
  constexpr size_t WK8_OFF  = 0x0C10000;
  constexpr size_t WV8_OFF  = 0x1010000;
  constexpr size_t WO8_OFF  = 0x1410000;
  constexpr size_t SWQ_OFF  = 0x1810000;
  constexpr size_t SWK_OFF  = 0x1820000;
  constexpr size_t SWV_OFF  = 0x1830000;
  constexpr size_t SWO_OFF  = 0x1840000;
  constexpr size_t QLIN_OFF = 0x1850000;
  constexpr size_t KLIN_OFF = 0x3850000;
  constexpr size_t VLIN_OFF = 0x5850000;
  constexpr size_t QH_OFF   = 0x7850000;
  constexpr size_t KH_OFF   = 0x8850000;
  constexpr size_t VH_OFF   = 0x9850000;

  int8_t* xq   = (int8_t*)(ws + XQ_OFF);
  float*  sx   = (float*)(ws + SX_OFF);
  int8_t* wq8  = (int8_t*)(ws + WQ8_OFF);
  int8_t* wk8  = (int8_t*)(ws + WK8_OFF);
  int8_t* wv8  = (int8_t*)(ws + WV8_OFF);
  int8_t* wo8  = (int8_t*)(ws + WO8_OFF);
  float*  swq  = (float*)(ws + SWQ_OFF);
  float*  swk  = (float*)(ws + SWK_OFF);
  float*  swv  = (float*)(ws + SWV_OFF);
  float*  swo  = (float*)(ws + SWO_OFF);
  float*  qlin = (float*)(ws + QLIN_OFF);
  float*  klin = (float*)(ws + KLIN_OFF);
  float*  vlin = (float*)(ws + VLIN_OFF);
  _Float16* qh = (_Float16*)(ws + QH_OFF);
  _Float16* kh = (_Float16*)(ws + KH_OFF);
  _Float16* vh = (_Float16*)(ws + VH_OFF);
  float*  attn_o = qlin;
  int8_t* oq     = xq;
  float*  so     = sx;

  k_quant_rows<<<M_, 256, 0, stream>>>(x,  xq,  sx,  C_);
  k_quant_rows<<<C_, 256, 0, stream>>>(wq, wq8, swq, C_);
  k_quant_rows<<<C_, 256, 0, stream>>>(wk, wk8, swk, C_);
  k_quant_rows<<<C_, 256, 0, stream>>>(wv, wv8, swv, C_);
  k_quant_rows<<<C_, 256, 0, stream>>>(wo, wo8, swo, C_);

  dim3 gg(M_ / 128, C_ / 32);
  k_gemm_q8<<<gg, 256, 0, stream>>>(xq, sx, wq8, swq, qlin);
  k_gemm_q8<<<gg, 256, 0, stream>>>(xq, sx, wk8, swk, klin);
  k_gemm_q8<<<gg, 256, 0, stream>>>(xq, sx, wv8, swv, vlin);

  k_norm_rope_cvt<<<dim3(M_, 3), 256, 0, stream>>>(
      qlin, klin, vlin, qg, kg, cosT, sinT, qh, kh, vh);

  k_attn<<<dim3(T_ / 64, H_, B_), 128, 0, stream>>>(qh, kh, vh, maskp, attn_o);

  k_quant_rows<<<M_, 256, 0, stream>>>(attn_o, oq, so, C_);
  k_gemm_q8<<<gg, 256, 0, stream>>>(oq, so, wo8, swo, (float*)d_out);
}
